// GumbelGroupVQ_8924942041843
// MI455X (gfx1250) — compile-verified, matrix operand order checked by reading, not run on hardware
//
#include <hip/hip_runtime.h>
#include <math.h>

// ---------------- Types for WMMA ----------------
typedef __attribute__((ext_vector_type(16))) __bf16 v16bf;
typedef __attribute__((ext_vector_type(8)))  float  v8f;

#define N_TOK   32768      // 8 * 4096
#define FEAT_I  512
#define HID     1024
#define CODE    640
#define CODE_G  320
#define FEAT_G  256
#define T_LEN   4096

// float -> bf16 (round to nearest even), stored as ushort
__device__ __forceinline__ unsigned short f2bf(float x) {
    unsigned int u = __float_as_uint(x);
    u += 0x7FFFu + ((u >> 16) & 1u);
    return (unsigned short)(u >> 16);
}

// Load a 16x32 bf16 fragment slice for one lane: two contiguous 16B loads.
// p must point at (row, k0 + ((lane>>4)<<3)) in a row-major bf16 matrix.
__device__ __forceinline__ v16bf load_frag(const unsigned short* p) {
    union { uint4 u[2]; v16bf v; } un;
    un.u[0] = *(const uint4*)(p);        // K+0..7  (or +8..15 for hi lanes)
    un.u[1] = *(const uint4*)(p + 16);   // K+16..23 (or +24..31)
    return un.v;
}

__device__ __forceinline__ v8f wmma_bf16(v16bf a, v16bf b, v8f c) {
    return __builtin_amdgcn_wmma_f32_16x16x32_bf16(
        /*neg_a=*/false, a, /*neg_b=*/false, b,
        /*c_mod=*/(short)0, c, /*reuse_a=*/false, /*reuse_b=*/false);
}

// ---------------- Kernel 1: series (B,512,T) -> X bf16 (N,512) ----------------
__global__ __launch_bounds__(256) void k_transpose_series(
    const float* __restrict__ src, unsigned short* __restrict__ dst) {
    __shared__ float tile[32][33];
    const int b  = blockIdx.z;
    const int t0 = blockIdx.x * 32, f0 = blockIdx.y * 32;
    const int tx = threadIdx.x, ty = threadIdx.y;
    const float* s = src + ((size_t)b * FEAT_I + f0) * T_LEN + t0;
#pragma unroll
    for (int i = 0; i < 4; ++i)
        tile[ty + i * 8][tx] = s[(size_t)(ty + i * 8) * T_LEN + tx];
    __syncthreads();
    unsigned short* d = dst + ((size_t)(b * T_LEN + t0)) * FEAT_I + f0;
#pragma unroll
    for (int i = 0; i < 4; ++i)
        d[(size_t)(ty + i * 8) * FEAT_I + tx] = f2bf(tile[tx][ty + i * 8]);
}

// ------------- Kernel 2: W (R,C) f32 -> WT (C,R) bf16 (N-major) -------------
__global__ __launch_bounds__(256) void k_transpose_w(
    const float* __restrict__ src, unsigned short* __restrict__ dst, int R, int C) {
    __shared__ float tile[32][33];
    const int r0 = blockIdx.y * 32, c0 = blockIdx.x * 32;
    const int tx = threadIdx.x, ty = threadIdx.y;
#pragma unroll
    for (int i = 0; i < 4; ++i)
        tile[ty + i * 8][tx] = src[(size_t)(r0 + ty + i * 8) * C + c0 + tx];
    __syncthreads();
#pragma unroll
    for (int i = 0; i < 4; ++i)
        dst[(size_t)(c0 + ty + i * 8) * R + r0 + tx] = f2bf(tile[tx][ty + i * 8]);
}

// -------- Kernel 3: H = relu(X @ W1 + b1) -> bf16 (N,1024); 128x64 tile --------
__global__ __launch_bounds__(256) void k_gemm1(
    const unsigned short* __restrict__ X, const unsigned short* __restrict__ W1T,
    const float* __restrict__ b1, unsigned short* __restrict__ H) {
    const int lane  = threadIdx.x & 31;
    const int wave  = threadIdx.x >> 5;
    const int rlane = lane & 15;
    const int klane = (lane >> 4) << 3;   // 0 or 8
    const int mloc  = (lane >> 4) << 3;   // C layout: M = r + mloc
    const int mW = blockIdx.x * 128 + (wave & 3) * 32;
    const int nW = blockIdx.y * 64  + (wave >> 2) * 32;

    v8f acc[2][2] = {{{}, {}}, {{}, {}}};
    for (int k0 = 0; k0 < FEAT_I; k0 += 32) {
        v16bf a[2], b[2];
#pragma unroll
        for (int i = 0; i < 2; ++i)
            a[i] = load_frag(X + (size_t)(mW + i * 16 + rlane) * FEAT_I + k0 + klane);
#pragma unroll
        for (int j = 0; j < 2; ++j)
            b[j] = load_frag(W1T + (size_t)(nW + j * 16 + rlane) * FEAT_I + k0 + klane);
#pragma unroll
        for (int i = 0; i < 2; ++i)
#pragma unroll
            for (int j = 0; j < 2; ++j)
                acc[i][j] = wmma_bf16(a[i], b[j], acc[i][j]);
    }
#pragma unroll
    for (int j = 0; j < 2; ++j) {
        const int col  = nW + j * 16 + rlane;
        const float bv = b1[col];
#pragma unroll
        for (int i = 0; i < 2; ++i)
#pragma unroll
            for (int r = 0; r < 8; ++r) {
                float v = acc[i][j][r] + bv;
                v = v > 0.0f ? v : 0.0f;
                H[(size_t)(mW + i * 16 + mloc + r) * HID + col] = f2bf(v);
            }
    }
}

// ---- Kernel 4: logits = H @ W2 + b2, + gumbel, argmax per group, gather cb ----
__global__ __launch_bounds__(256) void k_gemm2_vq(
    const unsigned short* __restrict__ H, const unsigned short* __restrict__ W2T,
    const float* __restrict__ b2, const float* __restrict__ gu,
    const float* __restrict__ cb, float* __restrict__ qout, int* __restrict__ idxout) {
    __shared__ float slog[16 * CODE];   // 40 KB of 320 KB LDS
    __shared__ int   sidx[32];          // [grp*16 + row]

    const int lane  = threadIdx.x & 31;
    const int wave  = threadIdx.x >> 5;
    const int rlane = lane & 15;
    const int klane = (lane >> 4) << 3;
    const int mloc  = (lane >> 4) << 3;
    const int mBase   = blockIdx.x * 16;
    const int colBase = wave * 80;      // 8 waves x 80 cols = 640

    v8f acc[5] = {{}, {}, {}, {}, {}};
    for (int k0 = 0; k0 < HID; k0 += 32) {
        v16bf a = load_frag(H + (size_t)(mBase + rlane) * HID + k0 + klane);
#pragma unroll
        for (int ct = 0; ct < 5; ++ct) {
            v16bf b = load_frag(W2T + (size_t)(colBase + ct * 16 + rlane) * HID + k0 + klane);
            acc[ct] = wmma_bf16(a, b, acc[ct]);
        }
    }
    // logits + bias + gumbel noise, scaled by 1/tau, into LDS
#pragma unroll
    for (int ct = 0; ct < 5; ++ct) {
        const int col  = colBase + ct * 16 + rlane;
        const float bv = b2[col];
#pragma unroll
        for (int r = 0; r < 8; ++r) {
            const int m = mloc + r;
            const int n = mBase + m;
            const float u  = gu[(size_t)n * CODE + col];
            const float gn = -__logf(-__logf(u + 1e-10f) + 1e-10f);
            slog[m * CODE + col] = (acc[ct][r] + bv + gn) * 0.5f;   // /TAU
        }
    }
    __syncthreads();
    // per-(row, group) argmax over 320 codes; first-max tie-break
    if (threadIdx.x < 32) {
        const int row = threadIdx.x & 15;
        const int grp = threadIdx.x >> 4;
        const float* p = &slog[row * CODE + grp * CODE_G];
        float best = p[0];
        int   bi   = 0;
        for (int c = 1; c < CODE_G; ++c) {
            float v = p[c];
            if (v > best) { best = v; bi = c; }
        }
        sidx[grp * 16 + row] = bi;
        idxout[(size_t)(mBase + row) * 2 + grp] = bi;
    }
    __syncthreads();
    // q_series[b, c, t] = codebook[g*320 + idx][f]; t contiguous across 16 lanes
    const int tx  = threadIdx.x & 15;   // row in tile -> t
    const int cyc = threadIdx.x >> 4;   // 0..15
    const int n   = mBase + tx;
    const int bb  = n >> 12;            // / 4096
    const int tt  = n & 4095;
    for (int c = cyc; c < 512; c += 16) {
        const int g = c >> 8, f = c & 255;
        const int id = sidx[g * 16 + tx];
        const float v = cb[((size_t)(g * CODE_G + id)) * FEAT_G + f];
        qout[((size_t)bb * 512 + c) * T_LEN + tt] = v;
    }
}

// --------------------------------- Launch ---------------------------------
extern "C" void kernel_launch(void* const* d_in, const int* in_sizes, int n_in,
                              void* d_out, int out_size, void* d_ws, size_t ws_size,
                              hipStream_t stream) {
    const float* series = (const float*)d_in[0];
    const float* gumbel = (const float*)d_in[1];
    const float* W1     = (const float*)d_in[2];
    const float* b1     = (const float*)d_in[3];
    const float* W2     = (const float*)d_in[4];
    const float* b2     = (const float*)d_in[5];
    const float* cb     = (const float*)d_in[6];

    float* qout  = (float*)d_out;                       // 8*512*4096 floats
    int*   idxout = (int*)(qout + (size_t)N_TOK * 512); // (B,T,2) int32

    char* ws = (char*)d_ws;
    unsigned short* XB  = (unsigned short*)(ws);                    // 32 MB
    unsigned short* W1T = (unsigned short*)(ws + 33554432);         // 1 MB
    unsigned short* W2T = (unsigned short*)(ws + 34603008);         // 1.25 MB
    unsigned short* HB  = (unsigned short*)(ws + 35913728);         // 64 MB

    k_transpose_series<<<dim3(T_LEN / 32, FEAT_I / 32, 8), dim3(32, 8), 0, stream>>>(series, XB);
    k_transpose_w<<<dim3(HID / 32, FEAT_I / 32), dim3(32, 8), 0, stream>>>(W1, W1T, FEAT_I, HID);
    k_transpose_w<<<dim3(CODE / 32, HID / 32),   dim3(32, 8), 0, stream>>>(W2, W2T, HID, CODE);
    k_gemm1<<<dim3(N_TOK / 128, HID / 64), 256, 0, stream>>>(XB, W1T, b1, HB);
    k_gemm2_vq<<<dim3(N_TOK / 16), 256, 0, stream>>>(HB, W2T, b2, gumbel, cb, qout, idxout);
}
